// MultiHeadQuery_selfattention_63127429317122
// MI455X (gfx1250) — matrix operand order repaired and checked by reading, not compiled, not run on hardware
//
#include <hip/hip_runtime.h>
#include <hip/hip_bf16.h>

// ---------------------------------------------------------------------------
// Types for CDNA5 WMMA (wave32, 16x16x32 bf16 -> f32)
// ---------------------------------------------------------------------------
typedef __attribute__((ext_vector_type(16))) __bf16 bf16x16;
typedef __attribute__((ext_vector_type(8)))  float  floatx8;
typedef __attribute__((ext_vector_type(4)))  unsigned int uintx4;

union ABFrag { bf16x16 v; uintx4 q[2]; };

__device__ __forceinline__ unsigned short f32_to_bf16(float f) {
  unsigned int u = __float_as_uint(f);
  u += 0x7FFFu + ((u >> 16) & 1u);   // round-to-nearest-even
  return (unsigned short)(u >> 16);
}

// 16x64 C strip of C = A * B^T (4 accumulators; A fragment reused across 4 B tiles).
// A:[16 x 32*ksteps] rows (lda), B_src:[64 x 32*ksteps] rows (ldb), both bf16
// row-major along k; rows 16B aligned, ld multiple of 8.
__device__ __forceinline__ void wmma_strip_nt4(const unsigned short* __restrict__ A, int lda,
                                               const unsigned short* __restrict__ B, int ldb,
                                               int ksteps, floatx8 acc[4])
{
  const int lane = threadIdx.x & 31;
  const int half = lane >> 4;    // k-chunk selector
  const int l15  = lane & 15;    // m (for A) / n (for B)
  const unsigned short* arow = A + l15 * lda;
  const unsigned short* brow = B + l15 * ldb;
  for (int ks = 0; ks < ksteps; ++ks) {
    const int k0 = ks << 5;
    ABFrag a;
    // A operand: elems 0..7 = A[m][k0+8c .. +7], elems 8..15 = A[m][k0+16+8c .. +7]
    a.q[0] = *(const uintx4*)(arow + k0 + 8 * half);
    a.q[1] = *(const uintx4*)(arow + k0 + 16 + 8 * half);
#pragma unroll
    for (int t = 0; t < 4; ++t) {
      const unsigned short* br = brow + t * 16 * ldb;
      ABFrag b;
      // B operand: elems 0..7 = B_src[n][k0+8c .. +7], elems 8..15 = B_src[n][k0+16+8c .. +7]
      b.q[0] = *(const uintx4*)(br + k0 + 8 * half);
      b.q[1] = *(const uintx4*)(br + k0 + 16 + 8 * half);
      acc[t] = __builtin_amdgcn_wmma_f32_16x16x32_bf16(false, a.v, false, b.v,
                                                       (short)0, acc[t], false, false);
    }
  }
}

#define ZACC4(acc) floatx8 acc[4] = { {0.f,0.f,0.f,0.f,0.f,0.f,0.f,0.f}, \
                                      {0.f,0.f,0.f,0.f,0.f,0.f,0.f,0.f}, \
                                      {0.f,0.f,0.f,0.f,0.f,0.f,0.f,0.f}, \
                                      {0.f,0.f,0.f,0.f,0.f,0.f,0.f,0.f} }

#define WPB 4  // waves per block for GEMM kernels (blockDim.x = 128)

// ---------------------------------------------------------------------------
// fp32 -> bf16 conversion
// ---------------------------------------------------------------------------
__global__ void f2bf_kernel(const float* __restrict__ x, unsigned short* __restrict__ y, int n) {
  int i = blockIdx.x * blockDim.x + threadIdx.x;
  if (i < n) y[i] = f32_to_bf16(x[i]);
}

// ---------------------------------------------------------------------------
// Projection: C = X * W^T, X:[4*L,512] bf16, W:[512,512] bf16 (row = out, col = in).
// Output split per head: outH[h][b][l][d] (bf16); optional transposed outT[h][b][d][l].
// ---------------------------------------------------------------------------
__global__ void proj_kernel(const unsigned short* __restrict__ X,
                            const unsigned short* __restrict__ W,
                            unsigned short* __restrict__ outH,
                            unsigned short* __restrict__ outT,
                            int L)
{
  const int ntN4 = 8;                       // 512/64
  const int rows = 4 * L;
  const int total = (rows / 16) * ntN4;
  int strip = blockIdx.x * (blockDim.x >> 5) + (threadIdx.x >> 5);
  if (strip >= total) return;
  int tm = strip / ntN4, tn = strip % ntN4;

  ZACC4(acc);
  wmma_strip_nt4(X + tm * 16 * 512, 512, W + tn * 64 * 512, 512, 16, acc);

  const int lane = threadIdx.x & 31, half = lane >> 4;
#pragma unroll
  for (int t = 0; t < 4; ++t) {
    const int n = tn * 64 + t * 16 + (lane & 15);
    const int h = n >> 6, d = n & 63;
#pragma unroll
    for (int r = 0; r < 8; ++r) {
      int row = tm * 16 + r + 8 * half;
      int b = row / L, l = row % L;
      unsigned short bv = f32_to_bf16(acc[t][r]);
      outH[(((h * 4 + b) * L) + l) * 64 + d] = bv;
      if (outT) outT[(((h * 4 + b) * 64) + d) * L + l] = bv;
    }
  }
}

// ---------------------------------------------------------------------------
// Scores: S[hb][m][n] = (Qh[hb] * Kh[hb]^T) / 8.  Qh:[32][Lq][64], Kh:[32][256][64].
// ---------------------------------------------------------------------------
__global__ void scores_kernel(const unsigned short* __restrict__ Qh,
                              const unsigned short* __restrict__ Kh,
                              float* __restrict__ S, int Lq)
{
  const int ntN4 = 4;                        // 256/64
  const int spb = (Lq / 16) * ntN4;          // strips per (h,b)
  int strip = blockIdx.x * (blockDim.x >> 5) + (threadIdx.x >> 5);
  if (strip >= 32 * spb) return;
  int hb = strip / spb, t0 = strip % spb;
  int tm = t0 / ntN4, tn = t0 % ntN4;

  ZACC4(acc);
  wmma_strip_nt4(Qh + (hb * Lq + tm * 16) * 64, 64,
                 Kh + (hb * 256 + tn * 64) * 64, 64, 2, acc);

  const int lane = threadIdx.x & 31, half = lane >> 4;
  float* Sb = S + (size_t)hb * Lq * 256;
#pragma unroll
  for (int t = 0; t < 4; ++t) {
    const int n0 = tn * 64 + t * 16 + (lane & 15);
#pragma unroll
    for (int r = 0; r < 8; ++r) {
      int row = tm * 16 + r + 8 * half;
      Sb[row * 256 + n0] = acc[t][r] * 0.125f;
    }
  }
}

// ---------------------------------------------------------------------------
// Row softmax over 256 cols; one wave per row. Writes bf16 and/or fp32.
// ---------------------------------------------------------------------------
__global__ void softmax_kernel(const float* __restrict__ S,
                               unsigned short* __restrict__ Pbf,
                               float* __restrict__ Pf32, int totalRows)
{
  int row = blockIdx.x * (blockDim.x >> 5) + (threadIdx.x >> 5);
  if (row >= totalRows) return;
  const int lane = threadIdx.x & 31;
  const float* sr = S + (size_t)row * 256;
  float v[8];
  float mx = -3.0e38f;
#pragma unroll
  for (int i = 0; i < 8; ++i) { v[i] = sr[lane + 32 * i]; mx = fmaxf(mx, v[i]); }
#pragma unroll
  for (int off = 16; off > 0; off >>= 1) mx = fmaxf(mx, __shfl_xor(mx, off));
  float sum = 0.f;
#pragma unroll
  for (int i = 0; i < 8; ++i) { v[i] = __expf(v[i] - mx); sum += v[i]; }
#pragma unroll
  for (int off = 16; off > 0; off >>= 1) sum += __shfl_xor(sum, off);
  float inv = 1.f / sum;
#pragma unroll
  for (int i = 0; i < 8; ++i) {
    float p = v[i] * inv;
    int c = lane + 32 * i;
    if (Pbf)  Pbf[(size_t)row * 256 + c]  = f32_to_bf16(p);
    if (Pf32) Pf32[(size_t)row * 256 + c] = p;
  }
}

// ---------------------------------------------------------------------------
// out = attn @ vh via vh^T.  attn:[32][256][256] bf16, vhT:[32][64][256] bf16.
// Each wave computes a full 16x64 output strip (one m-tile, all of dv).
// Stores concatenated per-head: outc[b][l][h*64+d] bf16 (hb = h*4+b).
// ---------------------------------------------------------------------------
__global__ void av_kernel(const unsigned short* __restrict__ attn,
                          const unsigned short* __restrict__ vhT,
                          unsigned short* __restrict__ outc)
{
  const int spb = 16;                      // 256/16 m-tiles, single 64-wide n-strip
  int strip = blockIdx.x * (blockDim.x >> 5) + (threadIdx.x >> 5);
  if (strip >= 32 * spb) return;
  int hb = strip / spb, tm = strip % spb;

  ZACC4(acc);
  wmma_strip_nt4(attn + ((size_t)hb * 256 + tm * 16) * 256, 256,
                 vhT  + ((size_t)hb * 64) * 256, 256, 8, acc);

  const int lane = threadIdx.x & 31, half = lane >> 4;
  const int h = hb >> 2, b = hb & 3;
#pragma unroll
  for (int t = 0; t < 4; ++t) {
    const int d = t * 16 + (lane & 15);
#pragma unroll
    for (int r = 0; r < 8; ++r) {
      int l = tm * 16 + r + 8 * half;
      outc[((size_t)(b * 256 + l)) * 512 + h * 64 + d] = f32_to_bf16(acc[t][r]);
    }
  }
}

// ---------------------------------------------------------------------------
// A2 = outc @ fc_w^T + fc_b + q   (fp32 out).  outc:[1024,512] bf16.
// ---------------------------------------------------------------------------
__global__ void a2_kernel(const unsigned short* __restrict__ outc,
                          const unsigned short* __restrict__ fcw,
                          const float* __restrict__ fcb,
                          const float* __restrict__ qres,
                          float* __restrict__ A2)
{
  const int ntN4 = 8;                      // 512/64
  const int total = 64 * ntN4;             // (1024/16) * 8
  int strip = blockIdx.x * (blockDim.x >> 5) + (threadIdx.x >> 5);
  if (strip >= total) return;
  int tm = strip / ntN4, tn = strip % ntN4;

  ZACC4(acc);
  wmma_strip_nt4(outc + tm * 16 * 512, 512, fcw + tn * 64 * 512, 512, 16, acc);

  const int lane = threadIdx.x & 31, half = lane >> 4;
#pragma unroll
  for (int t = 0; t < 4; ++t) {
    const int col = tn * 64 + t * 16 + (lane & 15);
    const float bias = fcb[col];
#pragma unroll
    for (int r = 0; r < 8; ++r) {
      int row = tm * 16 + r + 8 * half;
      A2[(size_t)row * 512 + col] = acc[t][r] + bias + qres[(size_t)row * 512 + col];
    }
  }
}

// ---------------------------------------------------------------------------
// Wv[h][b][k][m] = sum_d vh[h][b][k][d] * fc_w[m][64h+d]  (fp32 out).
// ---------------------------------------------------------------------------
__global__ void wv_kernel(const unsigned short* __restrict__ vh,
                          const unsigned short* __restrict__ fcw,
                          float* __restrict__ Wv)
{
  const int ntN4 = 8;                      // 512/64
  const int spb = 16 * ntN4;               // 128 strips per (h,b)
  int strip = blockIdx.x * (blockDim.x >> 5) + (threadIdx.x >> 5);
  if (strip >= 32 * spb) return;
  int hb = strip / spb, t0 = strip % spb;
  int tm = t0 / ntN4, tn = t0 % ntN4;
  const int h = hb >> 2;

  ZACC4(acc);
  wmma_strip_nt4(vh + ((size_t)hb * 256 + tm * 16) * 64, 64,
                 fcw + tn * 64 * 512 + 64 * h, 512, 2, acc);

  const int lane = threadIdx.x & 31, half = lane >> 4;
#pragma unroll
  for (int t = 0; t < 4; ++t) {
    const int col = tn * 64 + t * 16 + (lane & 15);
#pragma unroll
    for (int r = 0; r < 8; ++r) {
      int row = tm * 16 + r + 8 * half;
      Wv[((size_t)hb * 256 + row) * 512 + col] = acc[t][r];
    }
  }
}

// ---------------------------------------------------------------------------
// Combine + LayerNorm: res[b,n,k,:] = LN(A2[b,k,:] + sum_h qattn[h,b,n,k]*Wv[h,b,k,:])
// Block = 256 threads (8 waves) per (b,k). Wv tile cached in LDS, reused over n.
// ---------------------------------------------------------------------------
__global__ void combine_ln_kernel(const float* __restrict__ A2,     // [4][256][512]
                                  const float* __restrict__ qattn,  // [8][4][64][256]
                                  const float* __restrict__ Wv,     // [8][4][256][512]
                                  const float* __restrict__ lnw,
                                  const float* __restrict__ lnb,
                                  float* __restrict__ out)          // [4][64][256][512]
{
  __shared__ float sWv[8][512];
  __shared__ float sA2[512];
  __shared__ float slnw[512];
  __shared__ float slnb[512];

  const int bk = blockIdx.x;
  const int b = bk >> 8, k = bk & 255;
  const int tid = threadIdx.x;

  for (int i = tid; i < 512; i += 256) {
    sA2[i]  = A2[((size_t)(b * 256 + k)) * 512 + i];
    slnw[i] = lnw[i];
    slnb[i] = lnb[i];
#pragma unroll
    for (int h = 0; h < 8; ++h)
      sWv[h][i] = Wv[((size_t)((h * 4 + b) * 256 + k)) * 512 + i];
  }
  __syncthreads();

  const int wave = tid >> 5, lane = tid & 31;
  for (int n = wave; n < 64; n += 8) {
    float qa[8];
#pragma unroll
    for (int h = 0; h < 8; ++h)
      qa[h] = qattn[((size_t)((h * 4 + b) * 64 + n)) * 256 + k];

    float vals[16];
    float s = 0.f, s2 = 0.f;
#pragma unroll
    for (int i = 0; i < 16; ++i) {
      int m = lane + 32 * i;
      float x = sA2[m];
#pragma unroll
      for (int h = 0; h < 8; ++h) x += qa[h] * sWv[h][m];
      vals[i] = x; s += x; s2 += x * x;
    }
#pragma unroll
    for (int off = 16; off > 0; off >>= 1) {
      s  += __shfl_xor(s, off);
      s2 += __shfl_xor(s2, off);
    }
    const float mu  = s * (1.f / 512.f);
    const float var = s2 * (1.f / 512.f) - mu * mu;
    const float inv = rsqrtf(var + 1e-5f);
    float* orow = out + ((size_t)((b * 64 + n) * 256 + k)) * 512;
#pragma unroll
    for (int i = 0; i < 16; ++i) {
      int m = lane + 32 * i;
      orow[m] = (vals[i] - mu) * inv * slnw[m] + slnb[m];
    }
  }
}

// ---------------------------------------------------------------------------
// Host launch
// ---------------------------------------------------------------------------
extern "C" void kernel_launch(void* const* d_in, const int* in_sizes, int n_in,
                              void* d_out, int out_size, void* d_ws, size_t ws_size,
                              hipStream_t stream) {
  (void)in_sizes; (void)n_in; (void)out_size; (void)ws_size;

  const float* q     = (const float*)d_in[0];   // [4,256,512]
  const float* k     = (const float*)d_in[1];   // [4,256,512]
  const float* v     = (const float*)d_in[2];   // [4,256,512]
  const float* query = (const float*)d_in[3];   // [4,64,512]
  const float* w_qs  = (const float*)d_in[4];   // [512,512]
  const float* w_ks  = (const float*)d_in[5];
  const float* w_vs  = (const float*)d_in[6];
  const float* w_qry = (const float*)d_in[7];
  const float* fc_w  = (const float*)d_in[8];   // [512,512]
  const float* fc_b  = (const float*)d_in[9];   // [512]
  const float* ln_w  = (const float*)d_in[10];  // [512]
  const float* ln_b  = (const float*)d_in[11];  // [512]
  float* out = (float*)d_out;                   // [4,64,256,512]

  char* ws = (char*)d_ws;
  size_t o = 0;
  auto take = [&](size_t bytes) -> char* {
    char* p = ws + o;
    o = (o + bytes + 255) & ~(size_t)255;
    return p;
  };

  // persistent region
  unsigned short* qbf    = (unsigned short*)take(1024 * 512 * 2);
  unsigned short* kbf    = (unsigned short*)take(1024 * 512 * 2);
  unsigned short* vbf    = (unsigned short*)take(1024 * 512 * 2);
  unsigned short* qrybf  = (unsigned short*)take(256 * 512 * 2);
  unsigned short* wqsbf  = (unsigned short*)take(512 * 512 * 2);
  unsigned short* wksbf  = (unsigned short*)take(512 * 512 * 2);
  unsigned short* wvsbf  = (unsigned short*)take(512 * 512 * 2);
  unsigned short* wqrybf = (unsigned short*)take(512 * 512 * 2);
  unsigned short* fcwbf  = (unsigned short*)take(512 * 512 * 2);
  unsigned short* qh     = (unsigned short*)take(32 * 256 * 64 * 2);
  unsigned short* kh     = (unsigned short*)take(32 * 256 * 64 * 2);
  unsigned short* vh     = (unsigned short*)take(32 * 256 * 64 * 2);
  unsigned short* vhT    = (unsigned short*)take(32 * 64 * 256 * 2);
  unsigned short* qu     = (unsigned short*)take(32 * 64 * 64 * 2);
  unsigned short* attnbf = (unsigned short*)take(32 * 256 * 256 * 2);
  float*          qattn  = (float*)take(32 * 64 * 256 * 4);
  unsigned short* outc   = (unsigned short*)take(1024 * 512 * 2);
  float*          A2     = (float*)take(1024 * 512 * 4);

  // scratch region: scores (used first) aliased under Wv (used after softmax)
  size_t sbase = o;
  float* Sself = (float*)(ws + sbase);                       // 8 MiB
  float* Sqry  = (float*)(ws + sbase + 32u * 256 * 256 * 4); // 2 MiB
  float* Wv    = (float*)(ws + sbase);                       // 16 MiB, reuses score space

  // 1) fp32 -> bf16 conversions
  struct { const float* src; unsigned short* dst; int n; } cv[9] = {
    {q,     qbf,    1024 * 512}, {k,    kbf,    1024 * 512}, {v,     vbf,   1024 * 512},
    {query, qrybf,  256 * 512},  {w_qs, wqsbf,  512 * 512},  {w_ks,  wksbf, 512 * 512},
    {w_vs,  wvsbf,  512 * 512},  {w_qry, wqrybf, 512 * 512}, {fc_w,  fcwbf, 512 * 512},
  };
  for (int i = 0; i < 9; ++i)
    f2bf_kernel<<<(cv[i].n + 255) / 256, 256, 0, stream>>>(cv[i].src, cv[i].dst, cv[i].n);

  // 2) projections (strips/WPB blocks of 128 threads)
  proj_kernel<<<(64 * 8) / WPB, 128, 0, stream>>>(qbf,   wqsbf,  qh, nullptr, 256);
  proj_kernel<<<(64 * 8) / WPB, 128, 0, stream>>>(kbf,   wksbf,  kh, nullptr, 256);
  proj_kernel<<<(64 * 8) / WPB, 128, 0, stream>>>(vbf,   wvsbf,  vh, vhT,     256);
  proj_kernel<<<(16 * 8) / WPB, 128, 0, stream>>>(qrybf, wqrybf, qu, nullptr, 64);

  // 3) attention scores (scaled by 1/sqrt(64))
  scores_kernel<<<(32 * 16 * 4) / WPB, 128, 0, stream>>>(qh, kh, Sself, 256);
  scores_kernel<<<(32 * 4 * 4) / WPB, 128, 0, stream>>>(qu, kh, Sqry, 64);

  // 4) softmax
  softmax_kernel<<<(32 * 256) / WPB, 128, 0, stream>>>(Sself, attnbf, nullptr, 32 * 256);
  softmax_kernel<<<(32 * 64) / WPB, 128, 0, stream>>>(Sqry, nullptr, qattn, 32 * 64);

  // 5) out = attn @ vh  ->  outc[b][l][h*64+d]
  av_kernel<<<(32 * 16) / WPB, 128, 0, stream>>>(attnbf, vhT, outc);

  // 6) A2 = fc(outc) + fc_b + q ; Wv = per-head fc slice of vh (overwrites score scratch)
  a2_kernel<<<(64 * 8) / WPB, 128, 0, stream>>>(outc, fcwbf, fc_b, q, A2);
  wv_kernel<<<(32 * 16 * 8) / WPB, 128, 0, stream>>>(vh, fcwbf, Wv);

  // 7) combine + LayerNorm -> out
  combine_ln_kernel<<<4 * 256, 256, 0, stream>>>(A2, qattn, Wv, ln_w, ln_b, out);
}
